// SpikingMHA_16166256902444
// MI455X (gfx1250) — matrix operand order repaired and checked by reading, not compiled, not run on hardware
//
#include <hip/hip_runtime.h>
#include <hip/hip_bf16.h>

typedef __attribute__((ext_vector_type(16))) __bf16 v16bf;
typedef __attribute__((ext_vector_type(8)))  __bf16 v8bf;
typedef __attribute__((ext_vector_type(8)))  float  v8f;
typedef __attribute__((ext_vector_type(4)))  float  v4f;

#define EMBED 1024
#define NHEAD 16
#define HDIM  64
#define SEQ   2048
#define BATCH 2
#define NROWS (SEQ * BATCH)   // 4096 rows of x, flattened (t*B + b)
#define BHTOT (BATCH * NHEAD) // 32 (b,h) pairs

static __device__ __forceinline__ v8f wmma_bf16(v16bf a, v16bf b, v8f c) {
  // D = A(16x32 bf16) * B(32x16 bf16) + C(16x16 f32)
  return __builtin_amdgcn_wmma_f32_16x16x32_bf16(false, a, false, b,
                                                 (short)0, c, false, false);
}

static __device__ __forceinline__ v8bf pack_bf16(v4f w0, v4f w1) {
  v8bf wb;
#pragma unroll
  for (int i = 0; i < 4; i++) { wb[i] = (__bf16)w0[i]; wb[4 + i] = (__bf16)w1[i]; }
  return wb;
}

// ---------------------------------------------------------------------------
// Kernel A: y = x @ W^T + b for W in {Wq,Wk,Wv} (blockIdx.z selects).
// Writes Q,K as bf16 [bh][t][hd]; writes V transposed as Vt [bh][hd][t] so
// the attention kernel's P@V B-fragments are contiguous loads.
// Block = 256 threads = 8 waves; block tile = 128 rows x 64 cols; wave = 16x64.
// W tile is double-buffered in LDS: one barrier per 32-K step, and the next
// tile's global loads overlap the current step's WMMAs.
// ---------------------------------------------------------------------------
__global__ __launch_bounds__(256)
void spmha_qkv_kernel(const float* __restrict__ x,
                      const float* __restrict__ Wq, const float* __restrict__ bq,
                      const float* __restrict__ Wk, const float* __restrict__ bk,
                      const float* __restrict__ Wv, const float* __restrict__ bv,
                      __bf16* __restrict__ Qd, __bf16* __restrict__ Kd,
                      __bf16* __restrict__ Vt)
{
  __shared__ __align__(32) __bf16 ldsB[2][64 * 32];  // [buf][col][k], 2 x 4 KB

  const int which = blockIdx.z;                      // 0=Q 1=K 2=V
  const float* __restrict__ W    = (which == 0) ? Wq : (which == 1) ? Wk : Wv;
  const float* __restrict__ bias = (which == 0) ? bq : (which == 1) ? bk : bv;

  const int tid  = threadIdx.x;
  const int wave = tid >> 5;
  const int lane = tid & 31;
  const int l16  = lane & 15;
  const int hi   = lane >> 4;                        // 0: lanes 0-15, 1: lanes 16-31

  const int blk_m    = blockIdx.x * 128;
  const int col_base = blockIdx.y * 64;

  const float* __restrict__ xrow = x + (blk_m + wave * 16 + l16) * EMBED;

  // staging role of this thread for the W tile: [col][k] in LDS
  const int scol = tid >> 2;
  const int skp  = (tid & 3) * 8;
  const float* __restrict__ wbase = W + (col_base + scol) * EMBED + skp;

  v8f acc[4] = {};                                   // 16x64 f32 accumulators

  // prologue: stage tile for kk=0 into buffer 0
  {
    v4f w0 = *(const v4f*)(wbase);
    v4f w1 = *(const v4f*)(wbase + 4);
    *(v8bf*)&ldsB[0][scol * 32 + skp] = pack_bf16(w0, w1);
  }
  __syncthreads();

  for (int kk = 0; kk < EMBED; kk += 32) {
    const int  buf  = (kk >> 5) & 1;
    const bool more = (kk + 32) < EMBED;

    // ---- issue next W tile's global loads early (overlap with WMMAs) ----
    v4f w0, w1;
    if (more) {
      const float* wp = wbase + kk + 32;
      w0 = *(const v4f*)wp;
      w1 = *(const v4f*)(wp + 4);
      __builtin_prefetch(wp + 32, 0, 3);             // near-scope prefetch, tile after next
    }

    // ---- A fragment (16x32) from x, contiguous k = 16*hi + s, same order as B ----
    const int k0 = kk + hi * 16;
    v4f a0 = *(const v4f*)(xrow + k0);
    v4f a1 = *(const v4f*)(xrow + k0 + 4);
    v4f a2 = *(const v4f*)(xrow + k0 + 8);
    v4f a3 = *(const v4f*)(xrow + k0 + 12);
    v16bf af;
#pragma unroll
    for (int i = 0; i < 4; i++) {
      af[i]      = (__bf16)a0[i];
      af[4 + i]  = (__bf16)a1[i];
      af[8 + i]  = (__bf16)a2[i];
      af[12 + i] = (__bf16)a3[i];
    }

    // ---- 4 column tiles: B frag lane layout = [col=l16][k run by lane half] ----
#pragma unroll
    for (int j = 0; j < 4; j++) {
      v16bf bf = *(const v16bf*)&ldsB[buf][(j * 16 + l16) * 32 + hi * 16];
      acc[j] = wmma_bf16(af, bf, acc[j]);
    }

    // ---- store staged registers into the other buffer; one barrier/iter ----
    if (more)
      *(v8bf*)&ldsB[buf ^ 1][scol * 32 + skp] = pack_bf16(w0, w1);
    __syncthreads();
  }

  // ---- bias add + scatter store into attention-friendly layouts ----
#pragma unroll
  for (int j = 0; j < 4; j++) {
    const int e  = col_base + j * 16 + l16;          // output feature
    const int h  = e >> 6, hd = e & 63;
    const float bvv = bias[e];
#pragma unroll
    for (int r = 0; r < 8; r++) {
      const int row = blk_m + wave * 16 + r + hi * 8;  // flattened (t*B + b)
      const int t = row >> 1, bb = row & 1;
      const float y = acc[j][r] + bvv;
      if (which == 2) {
        Vt[((bb * NHEAD + h) * HDIM + hd) * SEQ + t] = (__bf16)y;
      } else {
        __bf16* O = which ? Kd : Qd;
        O[((bb * NHEAD + h) * SEQ + t) * HDIM + hd] = (__bf16)y;
      }
    }
  }
}

// ---------------------------------------------------------------------------
// Kernel B: per (b,h): S = QK^T/8 -> round(clip(.,0,4))/4 -> exp -> P@V / rowsum.
// Quantized scores are in [0,1] => softmax needs no max subtraction.
// Block = 8 waves; each wave owns 16 query rows, streams 2048 keys in 32-key
// tiles; P is re-laid-out C->A via a per-wave LDS bounce (same-wave DS is
// in-order, so no barriers in the key loop).
// ---------------------------------------------------------------------------
__global__ __launch_bounds__(256)
void spmha_attn_kernel(const __bf16* __restrict__ Qd,
                       const __bf16* __restrict__ Kd,
                       const __bf16* __restrict__ Vt,
                       float* __restrict__ out)
{
  __shared__ __align__(32) __bf16 ldsP[8 * 16 * 32];  // per-wave P staging, 8 KB

  const int tid  = threadIdx.x;
  const int wave = tid >> 5;
  const int lane = tid & 31;
  const int l16  = lane & 15;
  const int hi   = lane >> 4;

  const int bh = blockIdx.y;                       // 0..31
  const int bb = bh >> 4, h = bh & 15;
  const int q0 = blockIdx.x * 128 + wave * 16;     // first query row of this wave

  const __bf16* __restrict__ Qb = Qd + (size_t)bh * SEQ * HDIM;
  const __bf16* __restrict__ Kb = Kd + (size_t)bh * SEQ * HDIM;
  const __bf16* __restrict__ Vb = Vt + (size_t)bh * HDIM * SEQ;

  // ---- load Q as two A-frags (16 x 64 total) ----
  v16bf qf[2];
  const __bf16* qrow = Qb + (q0 + l16) * HDIM;
#pragma unroll
  for (int d = 0; d < 2; d++) {
    const int dk = d * 32 + hi * 16;
    v8bf r0 = *(const v8bf*)(qrow + dk);
    v8bf r1 = *(const v8bf*)(qrow + dk + 8);
#pragma unroll
    for (int i = 0; i < 8; i++) { qf[d][i] = r0[i]; qf[d][8 + i] = r1[i]; }
  }

  v8f o[4] = {};           // 16x64 output accumulator (stays in WMMA C)
  float dsum[8] = {};      // per-row partial sums of exp (per lane column-class)

  __bf16* myP = &ldsP[wave * 16 * 32];

  for (int s0 = 0; s0 < SEQ; s0 += 32) {
    // ---- S = Q @ K^T over 32 keys: 2 col tiles x 2 d-chunks = 4 WMMAs ----
    v8f sfrag[2] = {};
#pragma unroll
    for (int ct = 0; ct < 2; ct++) {
#pragma unroll
      for (int d = 0; d < 2; d++) {
        const __bf16* kp = Kb + (s0 + ct * 16 + l16) * HDIM + d * 32 + hi * 16;
        v16bf bf = *(const v16bf*)kp;              // contiguous 32B, B-frag ready
        sfrag[ct] = wmma_bf16(qf[d], bf, sfrag[ct]);
      }
    }

    // ---- spike quantize + exp; accumulate denominator; stage P in LDS ----
#pragma unroll
    for (int ct = 0; ct < 2; ct++) {
#pragma unroll
      for (int r = 0; r < 8; r++) {
        float s = sfrag[ct][r] * 0.125f;           // / sqrt(64)
        s = fminf(fmaxf(s, 0.0f), 4.0f);
        s = rintf(s) * 0.25f;                      // values in {0,.25,.5,.75,1}
        float p = __expf(s);                       // no max-subtraction needed
        dsum[r] += p;
        myP[(r + hi * 8) * 32 + ct * 16 + l16] = (__bf16)p;  // [row][key]
      }
    }
    // same-wave DS ops are in-order (DScnt); no barrier needed

    // ---- reload P as A-frag (16x32) ----
    v16bf pf;
    {
      const int dk = hi * 16;
      v8bf r0 = *(const v8bf*)(myP + l16 * 32 + dk);
      v8bf r1 = *(const v8bf*)(myP + l16 * 32 + dk + 8);
#pragma unroll
      for (int i = 0; i < 8; i++) { pf[i] = r0[i]; pf[8 + i] = r1[i]; }
    }

    // ---- O += P @ V : 4 hd-tiles, contiguous B-frags from Vt ----
#pragma unroll
    for (int j = 0; j < 4; j++) {
      const __bf16* vp = Vb + (j * 16 + l16) * SEQ + s0 + hi * 16;
      v16bf vf = *(const v16bf*)vp;                // contiguous 32B
      o[j] = wmma_bf16(pf, vf, o[j]);
    }
  }

  // ---- finish row sums: reduce across the 16-lane column group (wave32) ----
#pragma unroll
  for (int m = 1; m <= 8; m <<= 1) {
#pragma unroll
    for (int r = 0; r < 8; r++) dsum[r] += __shfl_xor(dsum[r], m, 32);
  }
  float inv[8];
#pragma unroll
  for (int r = 0; r < 8; r++) inv[r] = 1.0f / dsum[r];

  // ---- normalize and store fp32 output [T, B, D] ----
#pragma unroll
  for (int j = 0; j < 4; j++) {
    const int e = h * HDIM + j * 16 + l16;
#pragma unroll
    for (int r = 0; r < 8; r++) {
      const int t = q0 + r + hi * 8;
      out[(t * BATCH + bb) * EMBED + e] = o[j][r] * inv[r];
    }
  }
}

// ---------------------------------------------------------------------------
extern "C" void kernel_launch(void* const* d_in, const int* in_sizes, int n_in,
                              void* d_out, int out_size, void* d_ws, size_t ws_size,
                              hipStream_t stream) {
  const float* x  = (const float*)d_in[0];
  const float* Wq = (const float*)d_in[1];
  const float* bq = (const float*)d_in[2];
  const float* Wk = (const float*)d_in[3];
  const float* bk = (const float*)d_in[4];
  const float* Wv = (const float*)d_in[5];
  const float* bv = (const float*)d_in[6];

  const size_t qkv_elems = (size_t)BHTOT * SEQ * HDIM;  // 4,194,304 each
  __bf16* Qd = (__bf16*)d_ws;
  __bf16* Kd = Qd + qkv_elems;
  __bf16* Vt = Kd + qkv_elems;                          // 24 MB total

  dim3 gA(NROWS / 128, EMBED / 64, 3);                  // 32 x 16 x 3
  spmha_qkv_kernel<<<gA, 256, 0, stream>>>(x, Wq, bq, Wk, bk, Wv, bv, Qd, Kd, Vt);

  dim3 gB(SEQ / 128, BHTOT, 1);                         // 16 x 32
  spmha_attn_kernel<<<gB, 256, 0, stream>>>(Qd, Kd, Vt, (float*)d_out);
}
